// TopologicalGNN_3015067042089
// MI455X (gfx1250) — matrix operand order repaired and checked by reading, not compiled, not run on hardware
//
#include <hip/hip_runtime.h>

// ---------------- problem constants (from reference) ----------------
#define Bb   4
#define Nn   4096
#define Ee   131072
#define Hh   128
#define Ll   4
#define DIN  16
#define DE   8

typedef __attribute__((ext_vector_type(16))) _Float16 v16h;
typedef __attribute__((ext_vector_type(8)))  _Float16 h8;
typedef __attribute__((ext_vector_type(2)))  _Float16 h2;
typedef __attribute__((ext_vector_type(8)))  float    v8f;

union Frag { v16h v; h8 h[2]; };

__device__ __forceinline__ v8f wmma_f16(const Frag& a, const Frag& b, v8f c) {
  // D(16x16,f32) = A(16x32,f16) * B(32x16,f16) + C
  return __builtin_amdgcn_wmma_f32_16x16x32_f16(
      /*neg_a=*/false, a.v, /*neg_b=*/false, b.v,
      /*c_mod=*/(short)0, c, /*reuse_a=*/false, /*reuse_b=*/false);
}

__device__ __forceinline__ float silu(float x) {
  return x / (1.0f + __expf(-x));
}

// ---------------- weight packing: f32 (K x 128) -> f16 WMMA B-fragment order --
// out[((nt*KB+kb)*32 + lane)*16 + i] = W[k][n]
//   n = nt*16 + (lane&15);  k = kb*32 + (i<8 ? 8*(lane>>4)+i : 16+8*(lane>>4)+(i-8))
__global__ void pack_weight_kernel(const float* __restrict__ W,
                                   _Float16* __restrict__ out, int K) {
  const int l = blockIdx.y;                 // layer
  W   += (size_t)l * K * Hh;
  out += (size_t)l * K * Hh;
  const int KB = K >> 5;
  const int total = K * Hh;
  for (int o = blockIdx.x * blockDim.x + threadIdx.x; o < total;
       o += gridDim.x * blockDim.x) {
    int i    = o & 15;
    int lane = (o >> 4) & 31;
    int blk  = o >> 9;                      // nt*KB + kb
    int kb   = blk % KB;
    int nt   = blk / KB;
    int n = nt * 16 + (lane & 15);
    int k = kb * 32 + ((i < 8) ? (8 * (lane >> 4) + i) : (16 + 8 * (lane >> 4) + (i - 8)));
    out[o] = (_Float16)W[(size_t)k * Hh + n];
  }
}

// ---------------- node embedding: h = (h_in @ emb_W + b) * node_mask ---------
__global__ void embed_nodes_kernel(const float* __restrict__ hin,
                                   const float* __restrict__ W,
                                   const float* __restrict__ b,
                                   const float* __restrict__ node_mask,
                                   float* __restrict__ hout) {
  size_t idx = (size_t)blockIdx.x * blockDim.x + threadIdx.x;
  if (idx >= (size_t)Bb * Nn * Hh) return;
  size_t rn = idx >> 7;
  int c = (int)(idx & 127);
  const float* r = hin + rn * DIN;
  float acc = b[c];
#pragma unroll
  for (int k = 0; k < DIN; ++k) acc += r[k] * W[k * Hh + c];
  hout[idx] = acc * node_mask[rn];
}

// ---------------- edge embedding -> f16: ea = (ea_in @ W + b) * edge_mask ----
__global__ void embed_edges_kernel(const float* __restrict__ ein,
                                   const float* __restrict__ W,
                                   const float* __restrict__ b,
                                   const float* __restrict__ edge_mask,
                                   _Float16* __restrict__ eout) {
  size_t idx = (size_t)blockIdx.x * blockDim.x + threadIdx.x;
  if (idx >= (size_t)Bb * Ee * Hh) return;
  size_t rn = idx >> 7;
  int c = (int)(idx & 127);
  const float* r = ein + rn * DE;
  float acc = b[c];
#pragma unroll
  for (int k = 0; k < DE; ++k) acc += r[k] * W[k * Hh + c];
  eout[idx] = (_Float16)(acc * edge_mask[rn]);
}

// ---------------- edge MLP + scatter-add (WMMA core) ------------------------
// block = 256 threads (8 waves), one 32-edge tile.
// wave w owns N-slice w*16..w*16+15 and computes TWO 16-edge M-tiles,
// reusing each B fragment for 2 WMMAs (halves L2 weight traffic).
__global__ __launch_bounds__(256)
void edge_mlp_kernel(const float* __restrict__ hbuf,
                     const _Float16* __restrict__ ea16,
                     const int* __restrict__ edges,
                     const float* __restrict__ edge_mask,
                     const _Float16* __restrict__ W1p,  // 384x128 packed
                     const float* __restrict__ b1,
                     const _Float16* __restrict__ W2p,  // 128x128 packed
                     const float* __restrict__ b2,
                     float* __restrict__ agg) {
  constexpr int STR1 = 384 + 8;   // halves; 784 B row stride (16B multiple)
  constexpr int STR2 = 128 + 8;   // 272 B

  __shared__ __align__(16) _Float16 sEH[32 * STR1];
  __shared__ __align__(16) _Float16 sM1[32 * STR2];
  __shared__ int   sRow[32];
  __shared__ int   sCol[32];
  __shared__ float sEM[32];

  const int bi  = blockIdx.y;
  const int e0  = blockIdx.x << 5;
  const int tid = threadIdx.x;
  const int lane = tid & 31;
  const int wid  = tid >> 5;

  if (tid < 32) {
    size_t eg = (size_t)bi * Ee + (e0 + tid);
    sRow[tid] = edges[eg * 2 + 0];
    sCol[tid] = edges[eg * 2 + 1];
    sEM[tid]  = edge_mask[eg];
  }
  __syncthreads();

  // gather src/tgt rows + edge embedding into LDS as f16 (eh tile 32x384),
  // 2 columns per thread -> float2 global loads, packed h2 (b32) LDS stores
  for (int idx = tid; idx < 32 * 64; idx += 256) {
    int e = idx >> 6, cc = (idx & 63) << 1;
    float m = sEM[e];
    const float2 s = *(const float2*)(hbuf + ((size_t)bi * Nn + sRow[e]) * Hh + cc);
    const float2 t = *(const float2*)(hbuf + ((size_t)bi * Nn + sCol[e]) * Hh + cc);
    h2 ps; ps.x = (_Float16)(s.x * m); ps.y = (_Float16)(s.y * m);
    h2 pt; pt.x = (_Float16)(t.x * m); pt.y = (_Float16)(t.y * m);
    *(h2*)&sEH[e * STR1 + cc]          = ps;
    *(h2*)&sEH[e * STR1 + Hh + cc]     = pt;
    *(h2*)&sEH[e * STR1 + 2 * Hh + cc] =
        *(const h2*)&ea16[((size_t)bi * Ee + e0 + e) * Hh + cc];
  }
  __syncthreads();

  const int nt    = wid;            // this wave's 16-column output slice
  const int nlo   = lane & 15;
  const int hiSel = lane >> 4;      // 0: K {0-7,16-23}; 1: K {8-15,24-31}

  // ---- GEMM1: (32x384) @ (384x128 slice), K blocks = 12, B reused x2
  v8f acc0 = {}, acc1 = {};
  {
    const _Float16* bp  = W1p + ((size_t)(nt * 12) * 32 + lane) * 16;
    const _Float16* ap0 = sEH + nlo * STR1 + hiSel * 8;
    const _Float16* ap1 = ap0 + 16 * STR1;
#pragma unroll
    for (int kb = 0; kb < 12; ++kb) {
      Frag b;
      b.h[0] = *(const h8*)(bp);
      b.h[1] = *(const h8*)(bp + 8);
      bp += 32 * 16;
      Frag a0, a1;
      a0.h[0] = *(const h8*)(ap0 + kb * 32);
      a0.h[1] = *(const h8*)(ap0 + kb * 32 + 16);
      a1.h[0] = *(const h8*)(ap1 + kb * 32);
      a1.h[1] = *(const h8*)(ap1 + kb * 32 + 16);
      acc0 = wmma_f16(a0, b, acc0);
      acc1 = wmma_f16(a1, b, acc1);
    }
  }
  // bias + SiLU -> sM1 (f16). C layout: lane -> N, reg r -> M (+8 for hi lanes)
  {
    const float bias = b1[nt * 16 + nlo];
    const int mBase = hiSel * 8;
#pragma unroll
    for (int r = 0; r < 8; ++r) {
      sM1[(mBase + r) * STR2 + nt * 16 + nlo]      = (_Float16)silu(acc0[r] + bias);
      sM1[(16 + mBase + r) * STR2 + nt * 16 + nlo] = (_Float16)silu(acc1[r] + bias);
    }
  }
  __syncthreads();

  // ---- GEMM2: (32x128) @ (128x128 slice), K blocks = 4, B reused x2
  v8f acc2 = {}, acc3 = {};
  {
    const _Float16* bp  = W2p + ((size_t)(nt * 4) * 32 + lane) * 16;
    const _Float16* ap0 = sM1 + nlo * STR2 + hiSel * 8;
    const _Float16* ap1 = ap0 + 16 * STR2;
#pragma unroll
    for (int kb = 0; kb < 4; ++kb) {
      Frag b;
      b.h[0] = *(const h8*)(bp);
      b.h[1] = *(const h8*)(bp + 8);
      bp += 32 * 16;
      Frag a0, a1;
      a0.h[0] = *(const h8*)(ap0 + kb * 32);
      a0.h[1] = *(const h8*)(ap0 + kb * 32 + 16);
      a1.h[0] = *(const h8*)(ap1 + kb * 32);
      a1.h[1] = *(const h8*)(ap1 + kb * 32 + 16);
      acc2 = wmma_f16(a0, b, acc2);
      acc3 = wmma_f16(a1, b, acc3);
    }
  }
  // bias + SiLU + mask, scatter-add into agg[row[e]]
  {
    const float bias = b2[nt * 16 + nlo];
    const int mBase = hiSel * 8;
#pragma unroll
    for (int r = 0; r < 8; ++r) {
      int e = mBase + r;
      float y0 = silu(acc2[r] + bias) * sEM[e];
      float y1 = silu(acc3[r] + bias) * sEM[e + 16];
      float* d0 = &agg[((size_t)bi * Nn + sRow[e])      * Hh + nt * 16 + nlo];
      float* d1 = &agg[((size_t)bi * Nn + sRow[e + 16]) * Hh + nt * 16 + nlo];
      __hip_atomic_fetch_add(d0, y0, __ATOMIC_RELAXED, __HIP_MEMORY_SCOPE_AGENT);
      __hip_atomic_fetch_add(d1, y1, __ATOMIC_RELAXED, __HIP_MEMORY_SCOPE_AGENT);
    }
  }
}

// ---------------- node MLP: h += MLP([h, agg]); h *= node_mask --------------
// 32-node tiles, same double-M-tile B-reuse scheme
__global__ __launch_bounds__(256)
void node_mlp_kernel(float* __restrict__ hbuf,
                     const float* __restrict__ agg,
                     const float* __restrict__ node_mask,
                     const _Float16* __restrict__ W1p,  // 256x128 packed
                     const float* __restrict__ b1,
                     const _Float16* __restrict__ W2p,  // 128x128 packed
                     const float* __restrict__ b2) {
  constexpr int STRA = 256 + 8;   // 528 B
  constexpr int STR2 = 128 + 8;

  __shared__ __align__(16) _Float16 sNH[32 * STRA];
  __shared__ __align__(16) _Float16 sM1[32 * STR2];
  __shared__ float sNM[32];

  const int bi  = blockIdx.y;
  const int n0  = blockIdx.x << 5;
  const int tid = threadIdx.x;
  const int lane = tid & 31;
  const int wid  = tid >> 5;

  if (tid < 32) sNM[tid] = node_mask[(size_t)bi * Nn + n0 + tid];

  for (int idx = tid; idx < 32 * 64; idx += 256) {
    int r = idx >> 6, cc = (idx & 63) << 1;
    size_t g = ((size_t)bi * Nn + n0 + r) * Hh + cc;
    const float2 hv = *(const float2*)(hbuf + g);
    const float2 av = *(const float2*)(agg + g);
    h2 ph; ph.x = (_Float16)hv.x; ph.y = (_Float16)hv.y;
    h2 pa; pa.x = (_Float16)av.x; pa.y = (_Float16)av.y;
    *(h2*)&sNH[r * STRA + cc]      = ph;
    *(h2*)&sNH[r * STRA + Hh + cc] = pa;
  }
  __syncthreads();

  const int nt    = wid;
  const int nlo   = lane & 15;
  const int hiSel = lane >> 4;

  v8f acc0 = {}, acc1 = {};
  {
    const _Float16* bp  = W1p + ((size_t)(nt * 8) * 32 + lane) * 16;
    const _Float16* ap0 = sNH + nlo * STRA + hiSel * 8;
    const _Float16* ap1 = ap0 + 16 * STRA;
#pragma unroll
    for (int kb = 0; kb < 8; ++kb) {
      Frag b;
      b.h[0] = *(const h8*)(bp);
      b.h[1] = *(const h8*)(bp + 8);
      bp += 32 * 16;
      Frag a0, a1;
      a0.h[0] = *(const h8*)(ap0 + kb * 32);
      a0.h[1] = *(const h8*)(ap0 + kb * 32 + 16);
      a1.h[0] = *(const h8*)(ap1 + kb * 32);
      a1.h[1] = *(const h8*)(ap1 + kb * 32 + 16);
      acc0 = wmma_f16(a0, b, acc0);
      acc1 = wmma_f16(a1, b, acc1);
    }
  }
  {
    const float bias = b1[nt * 16 + nlo];
    const int mBase = hiSel * 8;
#pragma unroll
    for (int r = 0; r < 8; ++r) {
      sM1[(mBase + r) * STR2 + nt * 16 + nlo]      = (_Float16)silu(acc0[r] + bias);
      sM1[(16 + mBase + r) * STR2 + nt * 16 + nlo] = (_Float16)silu(acc1[r] + bias);
    }
  }
  __syncthreads();

  v8f acc2 = {}, acc3 = {};
  {
    const _Float16* bp  = W2p + ((size_t)(nt * 4) * 32 + lane) * 16;
    const _Float16* ap0 = sM1 + nlo * STR2 + hiSel * 8;
    const _Float16* ap1 = ap0 + 16 * STR2;
#pragma unroll
    for (int kb = 0; kb < 4; ++kb) {
      Frag b;
      b.h[0] = *(const h8*)(bp);
      b.h[1] = *(const h8*)(bp + 8);
      bp += 32 * 16;
      Frag a0, a1;
      a0.h[0] = *(const h8*)(ap0 + kb * 32);
      a0.h[1] = *(const h8*)(ap0 + kb * 32 + 16);
      a1.h[0] = *(const h8*)(ap1 + kb * 32);
      a1.h[1] = *(const h8*)(ap1 + kb * 32 + 16);
      acc2 = wmma_f16(a0, b, acc2);
      acc3 = wmma_f16(a1, b, acc3);
    }
  }
  {
    const float bias = b2[nt * 16 + nlo];
    const int mBase = hiSel * 8;
#pragma unroll
    for (int r = 0; r < 8; ++r) {
      int m = mBase + r;
      size_t g0 = ((size_t)bi * Nn + n0 + m)      * Hh + nt * 16 + nlo;
      size_t g1 = ((size_t)bi * Nn + n0 + m + 16) * Hh + nt * 16 + nlo;
      hbuf[g0] = (hbuf[g0] + acc2[r] + bias) * sNM[m];
      hbuf[g1] = (hbuf[g1] + acc3[r] + bias) * sNM[m + 16];
    }
  }
}

// ---------------- readout: sum_n softsign(h@gW1+gb1)@gW2 + gb2 -------------
__global__ __launch_bounds__(128)
void readout_kernel(const float* __restrict__ hbuf,
                    const float* __restrict__ node_mask,
                    const float* __restrict__ gW1, const float* __restrict__ gb1,
                    const float* __restrict__ gW2, const float* __restrict__ gb2,
                    float* __restrict__ out) {
  __shared__ float sh[Hh];
  __shared__ float sred[Hh];
  const size_t node = blockIdx.x;          // 0 .. B*N-1
  const int bi = (int)(node / Nn);
  const int c = threadIdx.x;

  sh[c] = hbuf[node * Hh + c];
  __syncthreads();

  float t = gb1[c];
#pragma unroll 8
  for (int k = 0; k < Hh; ++k) t += sh[k] * gW1[k * Hh + c];
  float ss = t / (1.0f + fabsf(t));
  sred[c] = ss * gW2[c];
  __syncthreads();
  for (int s = 64; s > 0; s >>= 1) {
    if (c < s) sred[c] += sred[c + s];
    __syncthreads();
  }
  if (c == 0) {
    float val = (sred[0] + gb2[0]) * node_mask[node];
    __hip_atomic_fetch_add(out + bi, val, __ATOMIC_RELAXED,
                           __HIP_MEMORY_SCOPE_AGENT);
  }
}

// ---------------- host side -------------------------------------------------
extern "C" void kernel_launch(void* const* d_in, const int* in_sizes, int n_in,
                              void* d_out, int out_size, void* d_ws, size_t ws_size,
                              hipStream_t stream) {
  const float* h_in      = (const float*)d_in[0];
  const int*   edges     = (const int*)d_in[1];
  const float* edge_attr = (const float*)d_in[2];
  const float* node_mask = (const float*)d_in[3];
  const float* edge_mask = (const float*)d_in[4];
  const float* emb_W  = (const float*)d_in[5];
  const float* emb_b  = (const float*)d_in[6];
  const float* eemb_W = (const float*)d_in[7];
  const float* eemb_b = (const float*)d_in[8];
  const float* eW1 = (const float*)d_in[9];
  const float* eb1 = (const float*)d_in[10];
  const float* eW2 = (const float*)d_in[11];
  const float* eb2 = (const float*)d_in[12];
  const float* nW1 = (const float*)d_in[13];
  const float* nb1 = (const float*)d_in[14];
  const float* nW2 = (const float*)d_in[15];
  const float* nb2 = (const float*)d_in[16];
  const float* gW1 = (const float*)d_in[17];
  const float* gb1 = (const float*)d_in[18];
  const float* gW2 = (const float*)d_in[19];
  const float* gb2 = (const float*)d_in[20];

  // workspace carve-up
  char* ws = (char*)d_ws;
  float* hbuf = (float*)ws;            ws += (size_t)Bb * Nn * Hh * sizeof(float);
  float* agg  = (float*)ws;            ws += (size_t)Bb * Nn * Hh * sizeof(float);
  _Float16* ea16 = (_Float16*)ws;      ws += (size_t)Bb * Ee * Hh * sizeof(_Float16);
  _Float16* W1p  = (_Float16*)ws;      ws += (size_t)Ll * 384 * Hh * sizeof(_Float16);
  _Float16* W2p  = (_Float16*)ws;      ws += (size_t)Ll * 128 * Hh * sizeof(_Float16);
  _Float16* nW1p = (_Float16*)ws;      ws += (size_t)Ll * 256 * Hh * sizeof(_Float16);
  _Float16* nW2p = (_Float16*)ws;      ws += (size_t)Ll * 128 * Hh * sizeof(_Float16);
  (void)in_sizes; (void)n_in; (void)ws_size;

  hipMemsetAsync(d_out, 0, (size_t)out_size * sizeof(float), stream);

  // pack all layer weights into WMMA B-fragment order (f16)
  pack_weight_kernel<<<dim3(192, Ll), 256, 0, stream>>>(eW1, W1p, 384);
  pack_weight_kernel<<<dim3(64,  Ll), 256, 0, stream>>>(eW2, W2p, 128);
  pack_weight_kernel<<<dim3(128, Ll), 256, 0, stream>>>(nW1, nW1p, 256);
  pack_weight_kernel<<<dim3(64,  Ll), 256, 0, stream>>>(nW2, nW2p, 128);

  // embeddings
  embed_nodes_kernel<<<(Bb * Nn * Hh) / 256, 256, 0, stream>>>(
      h_in, emb_W, emb_b, node_mask, hbuf);
  embed_edges_kernel<<<((size_t)Bb * Ee * Hh) / 256, 256, 0, stream>>>(
      edge_attr, eemb_W, eemb_b, edge_mask, ea16);

  for (int l = 0; l < Ll; ++l) {
    hipMemsetAsync(agg, 0, (size_t)Bb * Nn * Hh * sizeof(float), stream);
    edge_mlp_kernel<<<dim3(Ee / 32, Bb), 256, 0, stream>>>(
        hbuf, ea16, edges, edge_mask,
        W1p + (size_t)l * 384 * Hh, eb1 + l * Hh,
        W2p + (size_t)l * 128 * Hh, eb2 + l * Hh, agg);
    node_mlp_kernel<<<dim3(Nn / 32, Bb), 256, 0, stream>>>(
        hbuf, agg, node_mask,
        nW1p + (size_t)l * 256 * Hh, nb1 + l * Hh,
        nW2p + (size_t)l * 128 * Hh, nb2 + l * Hh);
  }

  readout_kernel<<<Bb * Nn, 128, 0, stream>>>(
      hbuf, node_mask, gW1, gb1, gW2, gb2, (float*)d_out);
}
